// EqPropTuned_47390669144411
// MI455X (gfx1250) — compile-verified, matrix operand order checked by reading, not run on hardware
//
#include <hip/hip_runtime.h>

// CDNA5 / gfx1250 EqProp relaxation.
// ~1.3 TFLOP of GEMM over a ~100MB working set -> L2-resident, compute bound.
// Strategy: bf16 WMMA (v_wmma_f32_16x16x32_bf16) with double-buffered LDS
// staged by async DMA (global_load_async_to_lds_b128, ASYNCcnt-tracked),
// BK=64 K-blocks so each barrier pair covers 16 WMMAs per wave, all dims
// zero-padded to multiples of 128 so the GEMM has no boundary guards, and a
// fused relaxation epilogue.

typedef __attribute__((ext_vector_type(16))) __bf16 v16bf;
typedef __attribute__((ext_vector_type(8)))  float  v8f;
typedef __attribute__((ext_vector_type(4)))  float  f4_t;

#define RELAX_LR 0.3f

// ---------------- helper kernels ----------------

// dst[r][c(dcols)] = bf16(src[r][c(scols)]) with zero-pad for c >= scols
__global__ void k_pad_convert(const float* __restrict__ src, __bf16* __restrict__ dst,
                              int rows, int scols, int dcols) {
  size_t i = (size_t)blockIdx.x * blockDim.x + threadIdx.x;
  size_t total = (size_t)rows * dcols;
  if (i >= total) return;
  int r = (int)(i / dcols), c = (int)(i % dcols);
  float v = (c < scols) ? src[(size_t)r * scols + c] : 0.0f;
  dst[i] = (__bf16)v;
}

// dst[n][k] = bf16(src[k][n]), dst is [drows x dcols], zero-pad out of range
__global__ void k_transpose_pad(const float* __restrict__ src, __bf16* __restrict__ dst,
                                int srows, int scols, int drows, int dcols) {
  size_t i = (size_t)blockIdx.x * blockDim.x + threadIdx.x;
  size_t total = (size_t)drows * dcols;
  if (i >= total) return;
  int n = (int)(i / dcols), k = (int)(i % dcols);
  float v = (n < scols && k < srows) ? src[(size_t)k * scols + n] : 0.0f;
  dst[i] = (__bf16)v;
}

__global__ void k_convert_clip(const float* __restrict__ src, __bf16* __restrict__ dst,
                               size_t n, int do_clip) {
  size_t i = (size_t)blockIdx.x * blockDim.x + threadIdx.x;
  if (i >= n) return;
  float v = src[i];
  if (do_clip) v = fminf(fmaxf(v, 0.0f), 1.0f);
  dst[i] = (__bf16)v;
}

__global__ void k_pad_bias(const float* __restrict__ src, float* __restrict__ dst,
                           int n, int npad) {
  int i = blockIdx.x * blockDim.x + threadIdx.x;
  if (i >= npad) return;
  dst[i] = (i < n) ? src[i] : 0.0f;
}

__global__ void k_extract(const float* __restrict__ s4, float* __restrict__ out) {
  size_t i = (size_t)blockIdx.x * blockDim.x + threadIdx.x;
  if (i >= (size_t)1024 * 1000) return;
  int m = (int)(i / 1000), n = (int)(i % 1000);
  out[i] = s4[(size_t)m * 1024 + n];
}

// ---------------- fused WMMA GEMM + relaxation ----------------
// C[m][n] = sum_k A1[m][k]*B1[n][k]  (+ sum_k A2[m][k]*B2[n][k] if K2>0)
// mode 0 (init):  S = clip(C + bias)
// mode 1 (relax): S = clip(0.7*S + 0.3*(bias + C))
// R = bf16(S)   (rho copy; states are in [0,1], so rho == identity)
// M = 1024 fixed; N multiple of 128; K1, K2 multiples of 64; B matrices are
// N-major (row n holds k-contiguous values), so A and B fragment loads match.
// Staging: double-buffered 128x64 LDS tiles filled by async DMA; the DMA for
// block t+1 overlaps the 16 WMMAs on block t; each wave waits its own
// ASYNCcnt before the workgroup barrier.
__launch_bounds__(256)
__global__ void k_eqprop_gemm(const __bf16* __restrict__ A1, const __bf16* __restrict__ B1, int K1,
                              const __bf16* __restrict__ A2, const __bf16* __restrict__ B2, int K2,
                              const float* __restrict__ bias,
                              float* __restrict__ S, __bf16* __restrict__ R,
                              int N, int mode) {
  __shared__ __bf16 lA[2][128 * 64];   // 2 x 16KB
  __shared__ __bf16 lB[2][128 * 64];   // 2 x 16KB

  const int tid  = threadIdx.x;
  const int lane = tid & 31;
  const int wave = tid >> 5;
  const int wm   = (wave & 3) * 32;   // wave's M offset in block tile (4 waves)
  const int wn   = (wave >> 2) * 64;  // wave's N offset in block tile (2 waves)
  const int bm   = blockIdx.y * 128;
  const int bn   = blockIdx.x * 128;
  const int lrow = lane & 15;         // row-within-tile for A/B fragments
  const int loff = (lane >> 4) * 8;   // k-chunk select (ISA 16-bit A layout)

  v8f acc[2][4] = {};                 // 8x 16x16 f32 accumulators

  const int n1 = K1 >> 6;             // #64-wide K blocks, phase 1
  const int n2 = K2 >> 6;             // phase 2 (0 if no transpose term)
  const int nt = n1 + n2;

  // Issue async DMA for iteration t into LDS buffer `buf`: this thread moves
  // four 16B chunks of A and four of B (256 thr * 4 * 16B = 16KB per tile).
  auto stage_async = [&](int t, int buf) {
    const __bf16* Ap; const __bf16* Bp; int K; int k0;
    if (t < n1) { Ap = A1; Bp = B1; K = K1; k0 = t * 64; }
    else        { Ap = A2; Bp = B2; K = K2; k0 = (t - n1) * 64; }
    unsigned la0 = (unsigned)(uintptr_t)(&lA[buf][0]);
    unsigned lb0 = (unsigned)(uintptr_t)(&lB[buf][0]);
#pragma unroll
    for (int it = 0; it < 4; ++it) {
      int c   = tid + it * 256;
      int row = c >> 3;           // 0..127
      int cg  = (c & 7) * 8;      // 0..56 halfs
      const __bf16* ga = Ap + (size_t)(bm + row) * K + k0 + cg;
      const __bf16* gb = Bp + (size_t)(bn + row) * K + k0 + cg;
      unsigned la = la0 + (unsigned)((row * 64 + cg) * 2);
      unsigned lb = lb0 + (unsigned)((row * 64 + cg) * 2);
      asm volatile("global_load_async_to_lds_b128 %0, %1, off"
                   :: "v"(la), "v"(ga) : "memory");
      asm volatile("global_load_async_to_lds_b128 %0, %1, off"
                   :: "v"(lb), "v"(gb) : "memory");
    }
  };

  // Consume LDS buffer `cur`: two 32-wide k sub-steps; per-lane fragments are
  // two 16B chunks at k = loff and k = 16 + loff (CDNA5 16-bit A layout).
  auto compute = [&](int cur) {
    const __bf16* baseA = &lA[cur][0];
    const __bf16* baseB = &lB[cur][0];
#pragma unroll
    for (int ks = 0; ks < 2; ++ks) {
      const int kk = ks * 32;
      union FragU { v16bf v; f4_t f[2]; } fa[2], fb[4];
#pragma unroll
      for (int tm = 0; tm < 2; ++tm) {
        const __bf16* p = baseA + (wm + tm * 16 + lrow) * 64 + kk;
        fa[tm].f[0] = *(const f4_t*)(p + loff);
        fa[tm].f[1] = *(const f4_t*)(p + 16 + loff);
      }
#pragma unroll
      for (int tn = 0; tn < 4; ++tn) {
        const __bf16* p = baseB + (wn + tn * 16 + lrow) * 64 + kk;
        fb[tn].f[0] = *(const f4_t*)(p + loff);
        fb[tn].f[1] = *(const f4_t*)(p + 16 + loff);
      }
#pragma unroll
      for (int tm = 0; tm < 2; ++tm)
#pragma unroll
        for (int tn = 0; tn < 4; ++tn)
          acc[tm][tn] = __builtin_amdgcn_wmma_f32_16x16x32_bf16(
              false, fa[tm].v, false, fb[tn].v, (short)0, acc[tm][tn],
              false, false);
    }
  };

  // Pipeline prologue: fill buffer 0, wave-local DMA wait, then WG barrier.
  stage_async(0, 0);
  asm volatile("s_wait_asynccnt 0" ::: "memory");
  __syncthreads();

  for (int t = 0; t < nt; ++t) {
    const int cur = t & 1;
    if (t + 1 < nt) stage_async(t + 1, cur ^ 1);  // DMA overlaps the WMMAs below
    compute(cur);
    if (t + 1 < nt) {
      asm volatile("s_wait_asynccnt 0" ::: "memory");
      __syncthreads();
    }
  }

  // Fused epilogue, uniform mode branch hoisted out of the element loops.
  // C/D layout: vgpr r, lanes 0-15 -> M=r, lanes 16-31 -> M=8+r.
  const int mhalf = (lane >> 4) * 8;
  if (mode == 0) {
#pragma unroll
    for (int tm = 0; tm < 2; ++tm)
#pragma unroll
      for (int tn = 0; tn < 4; ++tn) {
        const int n  = bn + wn + tn * 16 + lrow;
        const float bv = bias[n];
        const int m0 = bm + wm + tm * 16 + mhalf;
#pragma unroll
        for (int r = 0; r < 8; ++r) {
          size_t idx = (size_t)(m0 + r) * N + n;
          float v = fminf(fmaxf(acc[tm][tn][r] + bv, 0.0f), 1.0f);
          S[idx] = v;
          R[idx] = (__bf16)v;
        }
      }
  } else {
#pragma unroll
    for (int tm = 0; tm < 2; ++tm)
#pragma unroll
      for (int tn = 0; tn < 4; ++tn) {
        const int n  = bn + wn + tn * 16 + lrow;
        const float bv = bias[n];
        const int m0 = bm + wm + tm * 16 + mhalf;
#pragma unroll
        for (int r = 0; r < 8; ++r) {
          size_t idx = (size_t)(m0 + r) * N + n;
          float v = (1.0f - RELAX_LR) * S[idx] + RELAX_LR * (bv + acc[tm][tn][r]);
          v = fminf(fmaxf(v, 0.0f), 1.0f);
          S[idx] = v;
          R[idx] = (__bf16)v;
        }
      }
  }
}

// ---------------- host orchestration ----------------

extern "C" void kernel_launch(void* const* d_in, const int* in_sizes, int n_in,
                              void* d_out, int out_size, void* d_ws, size_t ws_size,
                              hipStream_t stream) {
  const float* x = (const float*)d_in[0];
  const float* W[4] = {(const float*)d_in[1], (const float*)d_in[2],
                       (const float*)d_in[3], (const float*)d_in[4]};
  const float* b[4] = {(const float*)d_in[5], (const float*)d_in[6],
                       (const float*)d_in[7], (const float*)d_in[8]};

  // Padded dims: D0..D3 = 2048, D4 = 1000 -> 1024.
  char* ws = (char*)d_ws;
  size_t off = 0;
  auto carve = [&](size_t bytes) -> char* {
    char* p = ws + off;
    off += (bytes + 255) & ~(size_t)255;
    return p;
  };

  __bf16* xb = (__bf16*)carve((size_t)1024 * 2048 * 2);  // bf16(x), unclipped
  __bf16* rx = (__bf16*)carve((size_t)1024 * 2048 * 2);  // bf16(rho(x))
  __bf16* r1 = (__bf16*)carve((size_t)1024 * 2048 * 2);
  __bf16* r2 = (__bf16*)carve((size_t)1024 * 2048 * 2);
  __bf16* r3 = (__bf16*)carve((size_t)1024 * 2048 * 2);
  __bf16* r4 = (__bf16*)carve((size_t)1024 * 1024 * 2);
  float*  s1 = (float*)carve((size_t)1024 * 2048 * 4);
  float*  s2 = (float*)carve((size_t)1024 * 2048 * 4);
  float*  s3 = (float*)carve((size_t)1024 * 2048 * 4);
  float*  s4 = (float*)carve((size_t)1024 * 1024 * 4);
  __bf16* Wb[4];   // row-major bf16 copies: B for the "@ W[l]^T" term
  __bf16* WT[4];   // transposed bf16 copies: B for the forward term
  Wb[0] = (__bf16*)carve((size_t)2048 * 2048 * 2);
  Wb[1] = (__bf16*)carve((size_t)2048 * 2048 * 2);
  Wb[2] = (__bf16*)carve((size_t)2048 * 2048 * 2);
  Wb[3] = (__bf16*)carve((size_t)2048 * 1024 * 2);
  WT[0] = (__bf16*)carve((size_t)2048 * 2048 * 2);
  WT[1] = (__bf16*)carve((size_t)2048 * 2048 * 2);
  WT[2] = (__bf16*)carve((size_t)2048 * 2048 * 2);
  WT[3] = (__bf16*)carve((size_t)1024 * 2048 * 2);
  float* b4p = (float*)carve((size_t)1024 * 4);

  const int T = 256;
  auto blocks = [](size_t n) { return (unsigned)((n + 255) / 256); };

  // Weight conversions (done every call; cheap vs. 1.3 TFLOP of GEMM).
  for (int i = 0; i < 3; ++i)
    k_pad_convert<<<blocks((size_t)2048 * 2048), T, 0, stream>>>(W[i], Wb[i], 2048, 2048, 2048);
  k_pad_convert<<<blocks((size_t)2048 * 1024), T, 0, stream>>>(W[3], Wb[3], 2048, 1000, 1024);
  for (int i = 0; i < 3; ++i)
    k_transpose_pad<<<blocks((size_t)2048 * 2048), T, 0, stream>>>(W[i], WT[i], 2048, 2048, 2048, 2048);
  k_transpose_pad<<<blocks((size_t)1024 * 2048), T, 0, stream>>>(W[3], WT[3], 2048, 1000, 1024, 2048);

  k_convert_clip<<<blocks((size_t)1024 * 2048), T, 0, stream>>>(x, xb, (size_t)1024 * 2048, 0);
  k_convert_clip<<<blocks((size_t)1024 * 2048), T, 0, stream>>>(x, rx, (size_t)1024 * 2048, 1);
  k_pad_bias<<<blocks(1024), T, 0, stream>>>(b[3], b4p, 1000, 1024);

  auto gemm = [&](const __bf16* A1, const __bf16* B1, int K1,
                  const __bf16* A2, const __bf16* B2, int K2,
                  const float* bias, float* S, __bf16* R, int N, int mode) {
    dim3 g(N / 128, 1024 / 128);
    k_eqprop_gemm<<<g, 256, 0, stream>>>(A1, B1, K1, A2, B2, K2, bias, S, R, N, mode);
  };

  // Feed-forward init: h_l = clip(h_{l-1} @ W[l-1] + b_l)  (first layer uses raw x)
  gemm(xb, WT[0], 2048, nullptr, nullptr, 0, b[0], s1, r1, 2048, 0);
  gemm(r1, WT[1], 2048, nullptr, nullptr, 0, b[1], s2, r2, 2048, 0);
  gemm(r2, WT[2], 2048, nullptr, nullptr, 0, b[2], s3, r3, 2048, 0);
  gemm(r3, WT[3], 2048, nullptr, nullptr, 0, b4p,  s4, r4, 1024, 0);

  // 25 Gauss-Seidel sweeps; launch order provides the in-sweep dependency.
  for (int it = 0; it < 25; ++it) {
    gemm(rx, WT[0], 2048, r2, Wb[1], 2048, b[0], s1, r1, 2048, 1);
    gemm(r1, WT[1], 2048, r3, Wb[2], 2048, b[1], s2, r2, 2048, 1);
    gemm(r2, WT[2], 2048, r4, Wb[3], 1024, b[2], s3, r3, 2048, 1);
    gemm(r3, WT[3], 2048, nullptr, nullptr, 0, b4p, s4, r4, 1024, 1);
  }

  k_extract<<<blocks((size_t)1024 * 1000), T, 0, stream>>>(s4, (float*)d_out);
}
